// NACF_52982716564136
// MI455X (gfx1250) — matrix operand order repaired and checked by reading, not compiled, not run on hardware
//
#include <hip/hip_runtime.h>
#include <hip/hip_bf16.h>

typedef __attribute__((ext_vector_type(16))) _Float16 v16h;
typedef __attribute__((ext_vector_type(8)))  float    v8f;

#define NB      16
#define TLEN    960000
#define STRIDE  160
#define WLEN    257
#define SFR     6001
#define TMAXV   213
#define KCAND   15
#define FLOORV  (-100000.0f)
#define C_OCT   0.01f
#define C_JUMP  0.35f
#define C_VUV   0.14f
#define TH_VOI  0.45f
#define TH_SIL  0.03f
#define FMINV   75.0f
#define FMAXV   600.0f

// ---------------- window normalized-ACF (once) ----------------
__global__ void prep_kernel(float* __restrict__ nacw) {
  __shared__ float w[WLEN];
  __shared__ float wsarr[TMAXV + 1];
  int t = threadIdx.x;
  for (int n = t; n < WLEN; n += 256)
    w[n] = 0.5f - 0.5f * cosf(6.283185307179586f * (float)n / (float)WLEN);
  __syncthreads();
  for (int tau = t; tau <= TMAXV; tau += 256) {
    float s = 0.f;
    for (int n = 0; n + tau < WLEN; ++n) s += w[n] * w[n + tau];
    wsarr[tau] = s;
  }
  __syncthreads();
  for (int tau = t; tau <= TMAXV; tau += 256) nacw[tau] = wsarr[tau] / wsarr[0];
}

// ---------------- per-batch global |x| peak ----------------
__global__ void peak_kernel(const float* __restrict__ x, float* __restrict__ gp) {
  int b = blockIdx.x, t = threadIdx.x;
  const float* xb = x + (size_t)b * TLEN;
  float m = 0.f;
  for (int i = t; i < TLEN; i += 256) m = fmaxf(m, fabsf(xb[i]));
  __shared__ float red[256];
  red[t] = m; __syncthreads();
  for (int o = 128; o > 0; o >>= 1) { if (t < o) red[t] = fmaxf(red[t], red[t + o]); __syncthreads(); }
  if (t == 0) gp[b] = red[0];
}

// ---------------- per-frame: WMMA ACF + candidates + top-15 ----------------
__global__ __launch_bounds__(32)
void frame_kernel(const float* __restrict__ x, const float* __restrict__ nacw,
                  const float* __restrict__ gpeak,
                  float* __restrict__ Freqs, float* __restrict__ Delta,
                  unsigned char* __restrict__ Voi) {
  const int lane = threadIdx.x;
  const int fid  = blockIdx.x;
  const int b = fid / SFR;
  const int s = fid % SFR;

  __shared__ float    fr[WLEN];      // raw frame, later reused for nacf
  __shared__ _Float16 F[544];        // zero-padded windowed frame (offset +16)
  __shared__ float    acf[240];      // [0..224] lags, [232] dump slot
  __shared__ float    cl[212];
  __shared__ float    cf[212];

  for (int i = lane; i < 544; i += 32) F[i] = (_Float16)0.f;

  // load frame + mean
  const size_t base = (size_t)b * TLEN + (size_t)s * STRIDE;
  float psum = 0.f;
  for (int n = lane; n < WLEN; n += 32) {
    float v = ((s * STRIDE + n) < TLEN) ? x[base + n] : 0.f;  // xpad zeros
    fr[n] = v; psum += v;
  }
  for (int o = 16; o > 0; o >>= 1) psum += __shfl_xor(psum, o);
  float mean = psum / (float)WLEN;
  __syncthreads();

  // window, store f16, local peak
  float lp = 0.f;
  for (int n = lane; n < WLEN; n += 32) {
    float wv = 0.5f - 0.5f * cosf(6.283185307179586f * (float)n / (float)WLEN);
    float wf = (fr[n] - mean) * wv;
    F[16 + n] = (_Float16)wf;
    lp = fmaxf(lp, fabsf(wf));
  }
  for (int o = 16; o > 0; o >>= 1) lp = fmaxf(lp, __shfl_xor(lp, o));
  float gp = gpeak[b];
  float intensity = (lp > gp) ? 1.f : lp / gp;
  __syncthreads();

  // ACF via chained v_wmma_f32_16x16x32_f16 (transposed form, all-contiguous reads):
  //   A'[M,k] = F[k0+k+16M]  (Hankel rows, per-lane contiguous -> ds_load_b128)
  //   B'[k,N] = F[k0+k+N]    (N = lane%16, per-lane contiguous -> ds_load_b128)
  //   D'[M][N] = acf(16M - N), unique (M,N) per lag
  v8f acc = {};
  const int m   = lane & 15;
  const int ghi = lane >> 4;
  for (int k0 = 0; k0 < 288; k0 += 32) {
    v16h afr, bfr;
    const int abase = k0 + 16 * m + 8 * ghi;
#pragma unroll
    for (int e = 0; e < 8; ++e)  afr[e] = F[abase + e];        // K = e + 8g
#pragma unroll
    for (int e = 8; e < 16; ++e) afr[e] = F[abase + 8 + e];    // K = 16 + (e-8) + 8g
    const int bbase = k0 + m + 8 * ghi;
#pragma unroll
    for (int e = 0; e < 8; ++e)  bfr[e] = F[bbase + e];        // col N = lane%16, K = e + 8g
#pragma unroll
    for (int e = 8; e < 16; ++e) bfr[e] = F[bbase + 8 + e];    // K = 16 + (e-8) + 8g
    acc = __builtin_amdgcn_wmma_f32_16x16x32_f16(false, afr, false, bfr,
                                                 (short)0, acc, false, false);
  }
  // scatter C/D tile: elem r -> M = r + 8*ghi, N = lane%16 ; lag = 16*M - N
  {
    int nn = lane & 15;
#pragma unroll
    for (int r = 0; r < 8; ++r) {
      int M = r + 8 * ghi;
      unsigned L = (unsigned)(16 * M - nn);
      unsigned idx = (L <= 224u) ? L : 232u;   // branchless: OOB lags -> dump slot
      acf[idx] = acc[r];
    }
  }
  __syncthreads();

  // nacf into fr[0..TMAXV]
  float a0 = acf[0];
  for (int tau = lane; tau <= TMAXV; tau += 32)
    fr[tau] = acf[tau] / (a0 * nacw[tau]);
  __syncthreads();

  // candidates m in [0,212): center lag = m+1
  for (int mm = lane; mm < 212; mm += 32) {
    float p_ = fr[mm], c_ = fr[mm + 1], n_ = fr[mm + 2];
    bool flag = ((c_ - p_) >= 0.f) && ((n_ - c_) <= 0.f) && (c_ > 0.5f * TH_VOI);
    float dr  = 0.5f * (n_ - p_) / (2.f * c_ - n_ - p_);
    float per = dr + (float)mm;
    per = per > 0.f ? per : 0.f;
    float freq = 16000.f / (1.f + per);
    float lg = c_;
    if (lg > 1.f) lg = 1.f / lg;
    lg -= C_OCT * log2f(FMINV / freq);
    if (!flag) lg = FLOORV;
    cl[mm] = lg; cf[mm] = freq;
  }
  __syncthreads();

  // top-15 (descending, first-index tie-break like jax top_k)
  float myL = 0.f, myF = 0.f;
  for (int k = 0; k < KCAND; ++k) {
    float bvv = -3.0e38f; int bi = 1 << 30;
    for (int mm = lane; mm < 212; mm += 32) {
      float v = cl[mm];
      if (v > bvv) { bvv = v; bi = mm; }
    }
    for (int o = 16; o > 0; o >>= 1) {
      float ov = __shfl_xor(bvv, o); int oi = __shfl_xor(bi, o);
      if (ov > bvv || (ov == bvv && oi < bi)) { bvv = ov; bi = oi; }
    }
    if (lane == k) { myL = bvv; myF = cf[bi]; }
    if (lane == 0) cl[bi] = -3.0e38f;
    __syncthreads();
  }

  float luv = 2.f - intensity * ((1.f + TH_VOI) / TH_SIL);
  luv = (luv > 0.f ? luv : 0.f) + TH_VOI;
  if (lane < KCAND) {
    bool v = (myL > FLOORV) && (myF < FMAXV);
    float dl = v ? (myL - C_OCT * log2f(FMAXV / myF)) : luv;
    size_t oi = ((size_t)b * SFR + s) * KCAND + lane;
    Freqs[oi] = myF; Delta[oi] = dl; Voi[oi] = v ? 1 : 0;
  }
}

// ---------------- Viterbi forward ----------------
__global__ void viterbi_fwd(const float* __restrict__ Freqs, const float* __restrict__ Delta,
                            const unsigned char* __restrict__ Voi,
                            unsigned char* __restrict__ Ptr, int* __restrict__ Last) {
  int b = blockIdx.x, t = threadIdx.x;
  __shared__ float val[KCAND];
  __shared__ float sc[KCAND * KCAND];
  int c = t / KCAND, p = t % KCAND;
  if (t < KCAND) val[t] = Delta[((size_t)b * SFR) * KCAND + t];
  __syncthreads();
  for (int s = 1; s < SFR; ++s) {
    if (t < KCAND * KCAND) {
      size_t ip = ((size_t)b * SFR + s - 1) * KCAND + p;
      size_t ic = ((size_t)b * SFR + s) * KCAND + c;
      int vp = Voi[ip], vc = Voi[ic];
      float tr;
      if (!vp && !vc)      tr = 0.f;
      else if (vp != vc)   tr = C_VUV;
      else                 tr = C_JUMP * fabsf(log2f(Freqs[ip] / Freqs[ic]));
      sc[t] = val[p] - tr + Delta[ic];
      // hide next step's L2 latency behind the reduce phase
      if (t < KCAND && s + 1 < SFR) {
        size_t in = ((size_t)b * SFR + s + 1) * KCAND + t;
        __builtin_prefetch((const void*)&Freqs[in], 0, 1);
        __builtin_prefetch((const void*)&Delta[in], 0, 1);
      }
    }
    __syncthreads();
    if (t < KCAND) {
      float best = sc[t * KCAND]; int bp = 0;
      for (int pp = 1; pp < KCAND; ++pp) {
        float v = sc[t * KCAND + pp];
        if (v > best) { best = v; bp = pp; }   // first-max on ties
      }
      Ptr[((size_t)b * SFR + s) * KCAND + t] = (unsigned char)bp;
      val[t] = best;
    }
    __syncthreads();
  }
  if (t == 0) {
    float best = val[0]; int bi = 0;
    for (int i = 1; i < KCAND; ++i) if (val[i] > best) { best = val[i]; bi = i; }
    Last[b] = bi;
  }
}

// ---------------- backtrack + f0 ----------------
__global__ void viterbi_bwd(const float* __restrict__ Freqs, const unsigned char* __restrict__ Voi,
                            const unsigned char* __restrict__ Ptr, const int* __restrict__ Last,
                            float* __restrict__ f0) {
  int b = threadIdx.x;
  if (b >= NB) return;
  int st = Last[b];
  for (int s = SFR - 1; s >= 0; --s) {
    size_t idx = ((size_t)b * SFR + s) * KCAND + st;
    f0[(size_t)b * SFR + s] = Voi[idx] ? Freqs[idx] : 0.f;
    if (s > 0) st = Ptr[((size_t)b * SFR + s) * KCAND + st];
  }
}

// ---------------- median-5 (edge padded) ----------------
__global__ void median_kernel(const float* __restrict__ f0, float* __restrict__ out) {
  int i = blockIdx.x * blockDim.x + threadIdx.x;
  if (i >= NB * SFR) return;
  int b = i / SFR, t = i % SFR;
  float v[5];
#pragma unroll
  for (int d = 0; d < 5; ++d) {
    int tt = t + d - 2; tt = tt < 0 ? 0 : (tt > SFR - 1 ? SFR - 1 : tt);
    v[d] = f0[(size_t)b * SFR + tt];
  }
#define SW(a, bb) { float lo = fminf(v[a], v[bb]); float hi = fmaxf(v[a], v[bb]); v[a] = lo; v[bb] = hi; }
  SW(0,1) SW(3,4) SW(2,4) SW(2,3) SW(1,4) SW(0,3) SW(0,2) SW(1,3) SW(1,2)
#undef SW
  out[i] = v[2];
}

extern "C" void kernel_launch(void* const* d_in, const int* in_sizes, int n_in,
                              void* d_out, int out_size, void* d_ws, size_t ws_size,
                              hipStream_t stream) {
  const float* x = (const float*)d_in[0];
  float* out = (float*)d_out;
  char* w = (char*)d_ws;
  auto al = [](size_t v) { return (v + 255) & ~(size_t)255; };
  size_t off = 0;
  const size_t NC = (size_t)NB * SFR * KCAND;
  float* nacw          = (float*)(w + off); off = al(off + (TMAXV + 1) * sizeof(float));
  float* gp            = (float*)(w + off); off = al(off + NB * sizeof(float));
  float* Freqs         = (float*)(w + off); off = al(off + NC * sizeof(float));
  float* Delta         = (float*)(w + off); off = al(off + NC * sizeof(float));
  unsigned char* Voi   = (unsigned char*)(w + off); off = al(off + NC);
  unsigned char* Ptr   = (unsigned char*)(w + off); off = al(off + NC);
  int* Last            = (int*)(w + off); off = al(off + NB * sizeof(int));
  float* f0            = (float*)(w + off); off = al(off + (size_t)NB * SFR * sizeof(float));

  prep_kernel<<<1, 256, 0, stream>>>(nacw);
  peak_kernel<<<NB, 256, 0, stream>>>(x, gp);
  frame_kernel<<<NB * SFR, 32, 0, stream>>>(x, nacw, gp, Freqs, Delta, Voi);
  viterbi_fwd<<<NB, 256, 0, stream>>>(Freqs, Delta, Voi, Ptr, Last);
  viterbi_bwd<<<1, 32, 0, stream>>>(Freqs, Voi, Ptr, Last, f0);
  median_kernel<<<(NB * SFR + 255) / 256, 256, 0, stream>>>(f0, out);
}
